// Encoder_4569845203210
// MI455X (gfx1250) — matrix operand order repaired and checked by reading, not compiled, not run on hardware
//
#include <hip/hip_runtime.h>
#include <stdint.h>
#include <stddef.h>

#define VOCAB  32000
#define EMB    256
#define UNITS  1024
#define BATCH  64
#define SEQ    512
#define NG     (3*UNITS)

typedef __attribute__((ext_vector_type(16))) __bf16       v16bf;
typedef __attribute__((ext_vector_type(8)))  float        v8f;
typedef __attribute__((ext_vector_type(8)))  unsigned int v8u;
typedef __attribute__((ext_vector_type(4)))  unsigned int v4u;

// round-to-nearest-even f32 -> bf16
__device__ __forceinline__ unsigned short bf16cvt(float f) {
  unsigned int a = __builtin_bit_cast(unsigned int, f);
  return (unsigned short)((a + 0x7FFFu + ((a >> 16) & 1u)) >> 16);
}

// ---- prep: W [K,N] f32 -> WT [N,K] bf16 (B-operand: 16 contiguous bf16 per lane load)
__global__ void transpose_bf16_kernel(const float* __restrict__ W,
                                      unsigned short* __restrict__ WT,
                                      int K, int N) {
  int i = blockIdx.x * blockDim.x + threadIdx.x;
  if (i >= K * N) return;
  int n = i / K;
  int k = i - n * K;
  WT[i] = bf16cvt(W[(size_t)k * N + n]);
}

// ---- prep: one-time embedding gather + bf16 convert: xeB[b][t][e] = bf16(emb[x[b,t]][e])
__global__ void gather_emb_kernel(const int* __restrict__ x,
                                  const float* __restrict__ emb,
                                  unsigned short* __restrict__ xeB) {
  size_t i = (size_t)blockIdx.x * blockDim.x + threadIdx.x;
  if (i >= (size_t)BATCH * SEQ * EMB) return;
  int e  = (int)(i & (EMB - 1));
  int bt = (int)(i >> 8);                 // EMB = 256 = 2^8; bt = b*SEQ + t (b-major, matches x)
  xeB[i] = bf16cvt(emb[(size_t)x[bt] * EMB + e]);
}

// ---- prep: h0 -> f32 + bf16 working buffers
__global__ void init_h_kernel(const float* __restrict__ h0,
                              float* __restrict__ hF,
                              unsigned short* __restrict__ hB) {
  int i = blockIdx.x * blockDim.x + threadIdx.x;
  if (i < BATCH * UNITS) { float v = h0[i]; hF[i] = v; hB[i] = bf16cvt(v); }
}

__global__ void copy_state_kernel(const float* __restrict__ h, float* __restrict__ out) {
  int i = blockIdx.x * blockDim.x + threadIdx.x;
  if (i < BATCH * UNITS) out[i] = h[i];
}

// load A-fragment (contiguous k order): two 16B chunks at k+16*half and k+16*half+8
__device__ __forceinline__ v16bf load_afrag(const unsigned short* __restrict__ row,
                                            int kc, int half) {
  v4u a0 = *(const v4u*)(row + kc + 16 * half);
  v4u a1 = *(const v4u*)(row + kc + 16 * half + 8);
  v8u araw;
  araw[0] = a0[0]; araw[1] = a0[1]; araw[2] = a0[2]; araw[3] = a0[3];
  araw[4] = a1[0]; araw[5] = a1[1]; araw[6] = a1[2]; araw[7] = a1[3];
  return __builtin_bit_cast(v16bf, araw);
}

// ---- one GRU timestep, fully fused:
//   [xe_t | h] (bf16) x [WxT|WhT] via v_wmma_f32_16x16x32_bf16, gates in registers.
// grid = 32 blocks x 256 thr = 256 waves = 4 m-tiles x 64 u-tiles.
__global__ __launch_bounds__(256) void gru_step_kernel(
    const unsigned short* __restrict__ xeB,    // [B][SEQ][EMB] bf16 (pre-gathered)
    const unsigned short* __restrict__ WxT,    // [3U][EMB]   bf16
    const unsigned short* __restrict__ WhT,    // [3U][UNITS] bf16
    const float*          __restrict__ b_i,
    const float*          __restrict__ b_r,
    const float*          __restrict__ hF_prev,
    const unsigned short* __restrict__ hB_prev,
    float*                __restrict__ hF_next,
    unsigned short*       __restrict__ hB_next,
    float*                __restrict__ ys,     // [B,T,U]
    int t)
{
  const int lane  = threadIdx.x & 31;
  const int wave  = blockIdx.x * (blockDim.x >> 5) + (threadIdx.x >> 5);
  const int mtile = wave & 3;         // batch tile (16 rows)
  const int utile = wave >> 2;        // unit tile (16 cols)
  const int half  = lane >> 4;        // hi/lo half-wave (K split: contiguous, k = 16*half + s)
  const int l16   = lane & 15;

  const int u  = utile * 16 + l16;    // this lane's output unit (N index)
  const int bA = mtile * 16 + l16;    // this lane's A-matrix row (batch index)

  // accumulators start at the biases (constant across the 8 M-slots of a lane)
  const float biasZ  = b_i[u]             + b_r[u];
  const float biasR  = b_i[UNITS + u]     + b_r[UNITS + u];
  const float biasXH = b_i[2 * UNITS + u];
  const float biasHH = b_r[2 * UNITS + u];
  v8f accZ  = {biasZ, biasZ, biasZ, biasZ, biasZ, biasZ, biasZ, biasZ};
  v8f accR  = {biasR, biasR, biasR, biasR, biasR, biasR, biasR, biasR};
  v8f accXH = {biasXH, biasXH, biasXH, biasXH, biasXH, biasXH, biasXH, biasXH};
  v8f accHH = {biasHH, biasHH, biasHH, biasHH, biasHH, biasHH, biasHH, biasHH};

  // B-operand row bases (WT is [N][K], 16 contiguous bf16 per lane per k-chunk)
  const unsigned short* WxZ = WxT + (size_t)u               * EMB;
  const unsigned short* WxR = WxT + (size_t)(UNITS + u)     * EMB;
  const unsigned short* WxH = WxT + (size_t)(2 * UNITS + u) * EMB;
  const unsigned short* WhZ = WhT + (size_t)u               * UNITS;
  const unsigned short* WhR = WhT + (size_t)(UNITS + u)     * UNITS;
  const unsigned short* WhH = WhT + (size_t)(2 * UNITS + u) * UNITS;

  // ---- K region 1: input projection, K = 0..255 (pre-gathered bf16 embedding row)
  const unsigned short* xrow = xeB + ((size_t)bA * SEQ + t) * EMB;
  #pragma unroll 4
  for (int kc = 0; kc < EMB; kc += 32) {
    v16bf af = load_afrag(xrow, kc, half);
    const int kb = kc + 16 * half;
    v16bf bz = __builtin_bit_cast(v16bf, *(const v8u*)(WxZ + kb));
    v16bf br = __builtin_bit_cast(v16bf, *(const v8u*)(WxR + kb));
    v16bf bh = __builtin_bit_cast(v16bf, *(const v8u*)(WxH + kb));
    accZ  = __builtin_amdgcn_wmma_f32_16x16x32_bf16(false, af, false, bz, (short)0, accZ,  false, false);
    accR  = __builtin_amdgcn_wmma_f32_16x16x32_bf16(false, af, false, br, (short)0, accR,  false, false);
    accXH = __builtin_amdgcn_wmma_f32_16x16x32_bf16(false, af, false, bh, (short)0, accXH, false, false);
  }

  // ---- K region 2: recurrent projection, K = 0..1023 (h already bf16)
  const unsigned short* hrow = hB_prev + (size_t)bA * UNITS;
  #pragma unroll 4
  for (int kc = 0; kc < UNITS; kc += 32) {
    v16bf af = load_afrag(hrow, kc, half);
    const int kb = kc + 16 * half;
    v16bf bz = __builtin_bit_cast(v16bf, *(const v8u*)(WhZ + kb));
    v16bf br = __builtin_bit_cast(v16bf, *(const v8u*)(WhR + kb));
    v16bf bh = __builtin_bit_cast(v16bf, *(const v8u*)(WhH + kb));
    accZ  = __builtin_amdgcn_wmma_f32_16x16x32_bf16(false, af, false, bz, (short)0, accZ,  false, false);
    accR  = __builtin_amdgcn_wmma_f32_16x16x32_bf16(false, af, false, br, (short)0, accR,  false, false);
    accHH = __builtin_amdgcn_wmma_f32_16x16x32_bf16(false, af, false, bh, (short)0, accHH, false, false);
  }

  // ---- gate nonlinearity in registers, D layout: lane -> N=u, VGPR v -> M = v + 8*half
  #pragma unroll
  for (int v = 0; v < 8; ++v) {
    const int b = mtile * 16 + v + 8 * half;
    const float z  = 1.0f / (1.0f + __expf(-accZ[v]));
    const float r  = 1.0f / (1.0f + __expf(-accR[v]));
    const float hc = tanhf(accXH[v] + r * accHH[v]);
    const float hp = hF_prev[(size_t)b * UNITS + u];
    const float hn = z * hp + (1.0f - z) * hc;
    hF_next[(size_t)b * UNITS + u] = hn;
    hB_next[(size_t)b * UNITS + u] = bf16cvt(hn);
    ys[(size_t)b * SEQ * UNITS + (size_t)t * UNITS + u] = hn;
  }
}

extern "C" void kernel_launch(void* const* d_in, const int* in_sizes, int n_in,
                              void* d_out, int out_size, void* d_ws, size_t ws_size,
                              hipStream_t stream) {
  const int*   x      = (const int*)d_in[0];
  const float* hidden = (const float*)d_in[1];
  const float* emb    = (const float*)d_in[2];
  const float* Wx     = (const float*)d_in[3];
  const float* Wh     = (const float*)d_in[4];
  const float* b_i    = (const float*)d_in[5];
  const float* b_r    = (const float*)d_in[6];
  float* out = (float*)d_out;                 // [B,T,U] then [B,U]

  // workspace layout (all offsets 32B-aligned): ~25 MB total
  unsigned short* WxT = (unsigned short*)d_ws;                   // 3072*256   bf16
  unsigned short* WhT = WxT + (size_t)NG * EMB;                  // 3072*1024  bf16
  unsigned short* xeB = WhT + (size_t)NG * UNITS;                // 64*512*256 bf16
  unsigned short* hB0 = xeB + (size_t)BATCH * SEQ * EMB;         // 64*1024    bf16
  unsigned short* hB1 = hB0 + (size_t)BATCH * UNITS;
  float*          hF0 = (float*)(hB1 + (size_t)BATCH * UNITS);   // 64*1024    f32
  float*          hF1 = hF0 + (size_t)BATCH * UNITS;

  // one-time prep: weight transpose+convert, embedding gather+convert, h0 init
  {
    int nWx = NG * EMB;
    transpose_bf16_kernel<<<(nWx + 255) / 256, 256, 0, stream>>>(Wx, WxT, EMB, NG);
    int nWh = NG * UNITS;
    transpose_bf16_kernel<<<(nWh + 255) / 256, 256, 0, stream>>>(Wh, WhT, UNITS, NG);
    size_t nXe = (size_t)BATCH * SEQ * EMB;
    gather_emb_kernel<<<(unsigned)((nXe + 255) / 256), 256, 0, stream>>>(x, emb, xeB);
    init_h_kernel<<<(BATCH * UNITS + 255) / 256, 256, 0, stream>>>(hidden, hF0, hB0);
  }

  // sequential scan: 512 dependent step launches (graph-captured), ping-pong h
  for (int t = 0; t < SEQ; ++t) {
    const float*          hFp = (t & 1) ? hF1 : hF0;
    const unsigned short* hBp = (t & 1) ? hB1 : hB0;
    float*                hFn = (t & 1) ? hF0 : hF1;
    unsigned short*       hBn = (t & 1) ? hB0 : hB1;
    gru_step_kernel<<<32, 256, 0, stream>>>(xeB, WxT, WhT, b_i, b_r,
                                            hFp, hBp, hFn, hBn, out, t);
  }

  // SEQ even -> last step (t=511, odd) wrote hF0
  copy_state_kernel<<<(BATCH * UNITS + 255) / 256, 256, 0, stream>>>(
      hF0, out + (size_t)BATCH * SEQ * UNITS);
}
